// AttentionBasedSampler_23897198035033
// MI455X (gfx1250) — matrix operand order repaired and checked by reading, not compiled, not run on hardware
//
#include <hip/hip_runtime.h>
#include <hip/hip_bf16.h>

#define BB 2
#define LL 768
#define DD 768
#define HH 12
#define FFD 2048
#define NLAYERS 6
#define HDD 64

typedef __bf16 bf16_t;
typedef __attribute__((ext_vector_type(16))) __bf16 v16bf;
typedef __attribute__((ext_vector_type(8)))  __bf16 v8bf;
typedef __attribute__((ext_vector_type(8)))  float  v8f;

__device__ inline bf16_t f2bf(float f) { return (bf16_t)f; }

__device__ inline v8f wmma_bf16(v16bf a, v16bf b, v8f c) {
    return __builtin_amdgcn_wmma_f32_16x16x32_bf16(false, a, false, b, (short)0, c, false, false);
}

// ---------------------------------------------------------------------------
// Batched GEMM: C[m,n] = alpha * sum_k A[m,k]*Bt[n,k] + bias[n] (+relu)(+res)
// A: bf16 row-major [M,K] (lda), Bt: bf16 [N,K] (ldb) == B transposed.
// Block = 256 threads = 8 waves; block tile 128x64; wave tile 16x64.
// B tile (64x32) staged to LDS via global_load_async_to_lds_b128 (ASYNCcnt),
// double buffered with fixed buffer roles on a 2-step unrolled K loop; A
// fragments ping-pong between two register sets (no moves, no hazard nops).
// All K used here are multiples of 64.
// ---------------------------------------------------------------------------
__global__ void __launch_bounds__(256)
gemm_bf16_wmma(const bf16_t* __restrict__ A, long long sAb, long long sAh, int lda,
               const bf16_t* __restrict__ Bt, long long sBb, long long sBh, int ldb,
               const float* __restrict__ bias,
               const float* __restrict__ residual,
               float* __restrict__ outF, bf16_t* __restrict__ outB,
               long long sCb, long long sCh, int ldc,
               int K, int batchH, float alpha, int relu)
{
    // 64 rows x 32 k-elems, padded to 40 elems (80B) per row: lane bank
    // stride = 20 banks -> conflict-free b128 reads. Two buffers.
    __shared__ bf16_t bsm[2][64 * 40];
    const uint32_t BUFB = 64 * 40 * 2;   // bytes per buffer

    int z  = blockIdx.z;
    int zb = z / batchH, zh = z % batchH;
    const bf16_t* Ab = A  + zb * sAb + zh * sAh;
    const bf16_t* Bb = Bt + zb * sBb + zh * sBh;
    long long coff   = zb * sCb + zh * sCh;

    int tid  = threadIdx.x;
    int lane = tid & 31;
    int wave = tid >> 5;          // 0..7 -> 8 row strips of 16
    int lm   = lane & 15;
    int hi   = lane >> 4;

    int m0 = blockIdx.y * 128 + wave * 16;
    int n0 = blockIdx.x * 64;

    // async staging: thread t copies 16B chunk (t&3) of B row (t>>2)
    int srow = tid >> 2, schunk = tid & 3;
    const bf16_t* sgp = Bb + (long long)(n0 + srow) * ldb + schunk * 8;
    uint32_t slds = (uint32_t)(uintptr_t)(&bsm[0][0]) + (uint32_t)(srow * 80 + schunk * 16);

    // A fragment base: lane lm holds row m0+lm; chunks at k = hi*8 and 16+hi*8
    const bf16_t* arow = Ab + (long long)(m0 + lm) * lda + hi * 8;

    union Frag { v16bf v; v8bf h[2]; };
    v8f acc[4] = {v8f{0.f}, v8f{0.f}, v8f{0.f}, v8f{0.f}};

    auto stage = [&](uint32_t bufOff, int k) {
        asm volatile("global_load_async_to_lds_b128 %0, %1, off"
                     :: "v"(slds + bufOff), "v"(sgp + k) : "memory");
    };
    auto prefA = [&](Frag& f, int k) {
        f.h[0] = *(const v8bf*)(arow + k);
        f.h[1] = *(const v8bf*)(arow + k + 16);
    };
    auto compute = [&](const bf16_t* bbase, const Frag& f) {
        // read all 4 B fragments first (one ds clause), then 4 WMMAs
        const bf16_t* bp = bbase + lm * 40 + hi * 8;
        Frag b0, b1, b2, b3;
        b0.h[0] = *(const v8bf*)(bp);
        b0.h[1] = *(const v8bf*)(bp + 16);
        b1.h[0] = *(const v8bf*)(bp + 16 * 40);
        b1.h[1] = *(const v8bf*)(bp + 16 * 40 + 16);
        b2.h[0] = *(const v8bf*)(bp + 32 * 40);
        b2.h[1] = *(const v8bf*)(bp + 32 * 40 + 16);
        b3.h[0] = *(const v8bf*)(bp + 48 * 40);
        b3.h[1] = *(const v8bf*)(bp + 48 * 40 + 16);
        acc[0] = wmma_bf16(f.v, b0.v, acc[0]);
        acc[1] = wmma_bf16(f.v, b1.v, acc[1]);
        acc[2] = wmma_bf16(f.v, b2.v, acc[2]);
        acc[3] = wmma_bf16(f.v, b3.v, acc[3]);
    };

    Frag r0, r1;
    stage(0, 0);            // buf0 <- k=0
    prefA(r0, 0);

    for (int k0 = 0; k0 < K; k0 += 64) {
        // buf0 staged (1 async in flight); barrier also proves buf1 consumed
        asm volatile("s_wait_asynccnt 0x0" ::: "memory");
        __syncthreads();
        stage(BUFB, k0 + 32);           // buf1 <- k0+32 (overlaps WMMAs)
        prefA(r1, k0 + 32);
        compute(&bsm[0][0], r0);

        asm volatile("s_wait_asynccnt 0x0" ::: "memory");
        __syncthreads();
        if (k0 + 64 < K) {              // uniform branch
            stage(0, k0 + 64);          // buf0 <- k0+64
            prefA(r0, k0 + 64);
        }
        compute(&bsm[1][0], r1);
    }

#pragma unroll
    for (int j = 0; j < 4; ++j) {
        int n = n0 + j * 16 + lm;
        float bval = bias ? bias[n] : 0.f;
#pragma unroll
        for (int r = 0; r < 8; ++r) {
            int m = m0 + hi * 8 + r;   // ISA 32-bit C/D layout
            float v = acc[j][r] * alpha + bval;
            if (relu) v = fmaxf(v, 0.f);
            long long idx = coff + (long long)m * ldc + n;
            if (residual) v += residual[idx];
            if (outF) outF[idx] = v;
            if (outB) outB[idx] = f2bf(v);
        }
    }
}

// ---------------------------------------------------------------------------
// Weight transpose + f32->bf16 convert: W[K,N] f32 -> Wt[N,K] bf16
// ---------------------------------------------------------------------------
__global__ void __launch_bounds__(256)
wtrans_kernel(const float* __restrict__ W, bf16_t* __restrict__ Wt, int K, int N)
{
    __shared__ float tile[32][33];
    int n0 = blockIdx.x * 32, k0 = blockIdx.y * 32;
    int tx = threadIdx.x & 31, ty = threadIdx.x >> 5;   // 32x8
    for (int r = ty; r < 32; r += 8)
        tile[r][tx] = W[(long long)(k0 + r) * N + (n0 + tx)];
    __syncthreads();
    for (int r = ty; r < 32; r += 8)
        Wt[(long long)(n0 + r) * K + (k0 + tx)] = f2bf(tile[tx][r]);
}

// ---------------------------------------------------------------------------
// LayerNorm over last dim W, affine, optional relu, bf16 out. 1 block/row.
// ---------------------------------------------------------------------------
__global__ void __launch_bounds__(256)
ln_to_bf16(const float* __restrict__ x, const float* __restrict__ g,
           const float* __restrict__ be, bf16_t* __restrict__ y, int W, int relu)
{
    long long row = blockIdx.x;
    const float* xr = x + row * W;
    bf16_t* yr = y + row * W;
    __shared__ float red[256];
    int tid = threadIdx.x;

    float s = 0.f;
    for (int i = tid; i < W; i += 256) s += xr[i];
    red[tid] = s; __syncthreads();
    for (int off = 128; off > 0; off >>= 1) {
        if (tid < off) red[tid] += red[tid + off];
        __syncthreads();
    }
    float mean = red[0] / W; __syncthreads();

    float vs = 0.f;
    for (int i = tid; i < W; i += 256) { float d = xr[i] - mean; vs += d * d; }
    red[tid] = vs; __syncthreads();
    for (int off = 128; off > 0; off >>= 1) {
        if (tid < off) red[tid] += red[tid + off];
        __syncthreads();
    }
    float rstd = rsqrtf(red[0] / W + 1e-5f);

    for (int i = tid; i < W; i += 256) {
        float o = (xr[i] - mean) * rstd * g[i] + be[i];
        if (relu) o = fmaxf(o, 0.f);
        yr[i] = f2bf(o);
    }
}

// ---------------------------------------------------------------------------
// Softmax over rows of [B,H,L,L] with structure bias sw[B,L,L] broadcast on H.
// scores already hold alpha*QK^T. Output bf16 probs. 1 block (256thr) / row.
// ---------------------------------------------------------------------------
__global__ void __launch_bounds__(256)
softmax_bf16(const float* __restrict__ scores, const float* __restrict__ sw,
             bf16_t* __restrict__ probs)
{
    int row = blockIdx.x;                 // ((b*H)+h)*L + q
    int q = row % LL;
    int b = row / (LL * HH);
    const float* srow = scores + (long long)row * LL;
    const float* wrow = sw + ((long long)b * LL + q) * LL;
    bf16_t* prow = probs + (long long)row * LL;
    int tid = threadIdx.x;
    __shared__ float red[256];

    float v0 = srow[tid]       + wrow[tid];
    float v1 = srow[tid + 256] + wrow[tid + 256];
    float v2 = srow[tid + 512] + wrow[tid + 512];

    float m = fmaxf(v0, fmaxf(v1, v2));
    red[tid] = m; __syncthreads();
    for (int off = 128; off > 0; off >>= 1) {
        if (tid < off) red[tid] = fmaxf(red[tid], red[tid + off]);
        __syncthreads();
    }
    m = red[0]; __syncthreads();

    float e0 = __expf(v0 - m), e1 = __expf(v1 - m), e2 = __expf(v2 - m);
    red[tid] = e0 + e1 + e2; __syncthreads();
    for (int off = 128; off > 0; off >>= 1) {
        if (tid < off) red[tid] += red[tid + off];
        __syncthreads();
    }
    float inv = 1.f / red[0];

    prow[tid]       = f2bf(e0 * inv);
    prow[tid + 256] = f2bf(e1 * inv);
    prow[tid + 512] = f2bf(e2 * inv);
}

// ---------------------------------------------------------------------------
// Split qkv f32 [B,L,3,H,HD] -> q,k bf16 [B,H,L,HD] and vT bf16 [B,H,HD,L]
// ---------------------------------------------------------------------------
__global__ void __launch_bounds__(256)
split_qkv(const float* __restrict__ qkv, bf16_t* __restrict__ q,
          bf16_t* __restrict__ k, bf16_t* __restrict__ vt)
{
    int idx = blockIdx.x * 256 + threadIdx.x;        // over B*L*D
    if (idx >= BB * LL * DD) return;
    int d = idx % HDD;
    int h = (idx / HDD) % HH;
    int l = (idx / DD) % LL;
    int b = idx / (LL * DD);
    const float* base = qkv + ((long long)(b * LL + l)) * (3 * DD) + h * HDD + d;
    long long qi = (((long long)b * HH + h) * LL + l) * HDD + d;
    long long vi = (((long long)b * HH + h) * HDD + d) * LL + l;
    q[qi]  = f2bf(base[0]);
    k[qi]  = f2bf(base[DD]);
    vt[vi] = f2bf(base[2 * DD]);
}

__global__ void __launch_bounds__(256)
convert_f32_bf16(const float* __restrict__ in, bf16_t* __restrict__ out, int n)
{
    int i = blockIdx.x * 256 + threadIdx.x;
    if (i < n) out[i] = f2bf(in[i]);
}

__global__ void __launch_bounds__(256)
copy_f32(const float* __restrict__ in, float* __restrict__ out, int n)
{
    int i = blockIdx.x * 256 + threadIdx.x;
    if (i < n) out[i] = in[i];
}

// ===========================================================================
extern "C" void kernel_launch(void* const* d_in, const int* in_sizes, int n_in,
                              void* d_out, int out_size, void* d_ws, size_t ws_size,
                              hipStream_t stream)
{
    const float* x_in  = (const float*)d_in[0];
    const float* si    = (const float*)d_in[1];
    const float* se_w1 = (const float*)d_in[2];
    const float* se_b1 = (const float*)d_in[3];
    const float* se_g  = (const float*)d_in[4];
    const float* se_be = (const float*)d_in[5];
    const float* se_w2 = (const float*)d_in[6];
    const float* se_b2 = (const float*)d_in[7];
    const float* qkv_w = (const float*)d_in[8];
    const float* qkv_b = (const float*)d_in[9];
    const float* sp_w  = (const float*)d_in[10];
    const float* sp_b  = (const float*)d_in[11];
    const float* op_w  = (const float*)d_in[12];
    const float* op_b  = (const float*)d_in[13];
    const float* n1_g  = (const float*)d_in[14];
    const float* n1_b  = (const float*)d_in[15];
    const float* n2_g  = (const float*)d_in[16];
    const float* n2_b  = (const float*)d_in[17];
    const float* ff_w1 = (const float*)d_in[18];
    const float* ff_b1 = (const float*)d_in[19];
    const float* ff_w2 = (const float*)d_in[20];
    const float* ff_b2 = (const float*)d_in[21];
    const float* out_w = (const float*)d_in[22];
    const float* out_b = (const float*)d_in[23];

    // ---- workspace carve-out ----
    char* p = (char*)d_ws;
    auto alloc = [&](size_t bytes) -> void* {
        void* r = (void*)p;
        p += (bytes + 255) & ~(size_t)255;
        return r;
    };
    const int ROWS = BB * LL;                 // 1536
    bf16_t* wt_se1 = (bf16_t*)alloc((size_t)FFD * DD * 2);
    bf16_t* wt_se2 = (bf16_t*)alloc((size_t)DD * FFD * 2);
    bf16_t* wt_out = (bf16_t*)alloc((size_t)DD * DD * 2);
    bf16_t* wt_qkv = (bf16_t*)alloc((size_t)NLAYERS * 3 * DD * DD * 2);
    bf16_t* wt_sp  = (bf16_t*)alloc((size_t)NLAYERS * DD * DD * 2);
    bf16_t* wt_op  = (bf16_t*)alloc((size_t)NLAYERS * DD * DD * 2);
    bf16_t* wt_f1  = (bf16_t*)alloc((size_t)NLAYERS * FFD * DD * 2);
    bf16_t* wt_f2  = (bf16_t*)alloc((size_t)NLAYERS * DD * FFD * 2);
    float*  xbuf   = (float*)alloc((size_t)ROWS * DD * 4);
    bf16_t* xn_bf  = (bf16_t*)alloc((size_t)ROWS * DD * 2);
    float*  qkvf   = (float*)alloc((size_t)ROWS * 3 * DD * 4);
    bf16_t* q_bf   = (bf16_t*)alloc((size_t)ROWS * DD * 2);
    bf16_t* k_bf   = (bf16_t*)alloc((size_t)ROWS * DD * 2);
    bf16_t* vt_bf  = (bf16_t*)alloc((size_t)ROWS * DD * 2);
    float*  scores = (float*)alloc((size_t)BB * HH * LL * LL * 4);
    bf16_t* probs  = (bf16_t*)alloc((size_t)BB * HH * LL * LL * 2);
    float*  swf    = (float*)alloc((size_t)BB * LL * LL * 4);
    bf16_t* o_bf   = (bf16_t*)alloc((size_t)ROWS * DD * 2);
    float*  h_f32  = (float*)alloc((size_t)ROWS * FFD * 4);
    bf16_t* h_bf   = (bf16_t*)alloc((size_t)ROWS * FFD * 2);
    bf16_t* sb_bf  = (bf16_t*)alloc((size_t)ROWS * DD * 2);
    bf16_t* si_bf  = (bf16_t*)alloc((size_t)ROWS * DD * 2);

    auto wtrans = [&](const float* W, bf16_t* Wt, int K, int N) {
        wtrans_kernel<<<dim3(N / 32, K / 32), 256, 0, stream>>>(W, Wt, K, N);
    };
    auto gemm = [&](const bf16_t* A, int lda, const bf16_t* Bt, int ldb,
                    const float* bias, const float* res,
                    float* oF, bf16_t* oB, int ldc,
                    int M, int N, int K, float alpha, int relu) {
        gemm_bf16_wmma<<<dim3(N / 64, M / 128, 1), 256, 0, stream>>>(
            A, 0, 0, lda, Bt, 0, 0, ldb, bias, res, oF, oB, 0, 0, ldc,
            K, 1, alpha, relu);
    };

    // ---- weight prep (every call; deterministic) ----
    wtrans(se_w1, wt_se1, DD, FFD);
    wtrans(se_w2, wt_se2, FFD, DD);
    wtrans(out_w, wt_out, DD, DD);
    for (int i = 0; i < NLAYERS; ++i) {
        wtrans(qkv_w + (size_t)i * DD * 3 * DD, wt_qkv + (size_t)i * 3 * DD * DD, DD, 3 * DD);
        wtrans(sp_w  + (size_t)i * DD * DD,     wt_sp  + (size_t)i * DD * DD,     DD, DD);
        wtrans(op_w  + (size_t)i * DD * DD,     wt_op  + (size_t)i * DD * DD,     DD, DD);
        wtrans(ff_w1 + (size_t)i * DD * FFD,    wt_f1  + (size_t)i * FFD * DD,    DD, FFD);
        wtrans(ff_w2 + (size_t)i * FFD * DD,    wt_f2  + (size_t)i * DD * FFD,    FFD, DD);
    }

    const int NELEM = ROWS * DD;
    copy_f32<<<(NELEM + 255) / 256, 256, 0, stream>>>(x_in, xbuf, NELEM);
    convert_f32_bf16<<<(NELEM + 255) / 256, 256, 0, stream>>>(si, si_bf, NELEM);

    // ---- structure encoder ----
    gemm(si_bf, DD, wt_se1, DD, se_b1, nullptr, h_f32, nullptr, FFD,
         ROWS, FFD, DD, 1.f, 0);
    ln_to_bf16<<<ROWS, 256, 0, stream>>>(h_f32, se_g, se_be, h_bf, FFD, 1);
    gemm(h_bf, FFD, wt_se2, FFD, se_b2, nullptr, nullptr, sb_bf, DD,
         ROWS, DD, FFD, 1.f, 0);

    const float scale = 0.125f;  // 1/sqrt(64)
    for (int i = 0; i < NLAYERS; ++i) {
        ln_to_bf16<<<ROWS, 256, 0, stream>>>(xbuf, n1_g + i * DD, n1_b + i * DD, xn_bf, DD, 0);
        gemm(xn_bf, DD, wt_qkv + (size_t)i * 3 * DD * DD, DD, qkv_b + (size_t)i * 3 * DD,
             nullptr, qkvf, nullptr, 3 * DD, ROWS, 3 * DD, DD, 1.f, 0);
        split_qkv<<<(NELEM + 255) / 256, 256, 0, stream>>>(qkvf, q_bf, k_bf, vt_bf);
        // scores = scale * Q K^T   (batched over B*H)
        gemm_bf16_wmma<<<dim3(LL / 64, LL / 128, BB * HH), 256, 0, stream>>>(
            q_bf, (long long)HH * LL * HDD, (long long)LL * HDD, HDD,
            k_bf, (long long)HH * LL * HDD, (long long)LL * HDD, HDD,
            nullptr, nullptr, scores, nullptr,
            (long long)HH * LL * LL, (long long)LL * LL, LL,
            HDD, HH, scale, 0);
        // sw = sb @ sp_w + sp_b
        gemm(sb_bf, DD, wt_sp + (size_t)i * DD * DD, DD, sp_b + (size_t)i * DD,
             nullptr, swf, nullptr, DD, ROWS, DD, DD, 1.f, 0);
        softmax_bf16<<<BB * HH * LL, 256, 0, stream>>>(scores, swf, probs);
        // o = P @ V  (batched; writes bf16 [B,L,D] with head interleave)
        gemm_bf16_wmma<<<dim3(HDD / 64, LL / 128, BB * HH), 256, 0, stream>>>(
            probs, (long long)HH * LL * LL, (long long)LL * LL, LL,
            vt_bf, (long long)HH * HDD * LL, (long long)HDD * LL, LL,
            nullptr, nullptr, nullptr, o_bf,
            (long long)LL * DD, (long long)HDD, DD,
            LL, HH, 1.f, 0);
        // x = x + o @ op_w + op_b
        gemm(o_bf, DD, wt_op + (size_t)i * DD * DD, DD, op_b + (size_t)i * DD,
             xbuf, xbuf, nullptr, DD, ROWS, DD, DD, 1.f, 0);
        // FFN
        ln_to_bf16<<<ROWS, 256, 0, stream>>>(xbuf, n2_g + i * DD, n2_b + i * DD, xn_bf, DD, 0);
        gemm(xn_bf, DD, wt_f1 + (size_t)i * FFD * DD, DD, ff_b1 + (size_t)i * FFD,
             nullptr, nullptr, h_bf, FFD, ROWS, FFD, DD, 1.f, 1);
        gemm(h_bf, FFD, wt_f2 + (size_t)i * DD * FFD, FFD, ff_b2 + (size_t)i * DD,
             xbuf, xbuf, nullptr, DD, ROWS, DD, FFD, 1.f, 0);
    }

    // ---- final projection ----
    convert_f32_bf16<<<(NELEM + 255) / 256, 256, 0, stream>>>(xbuf, xn_bf, NELEM);
    gemm(xn_bf, DD, wt_out, DD, out_b, nullptr, (float*)d_out, nullptr, DD,
         ROWS, DD, DD, 1.f, 0);
}
